// SNNModelSimple_90520730730913
// MI455X (gfx1250) — matrix operand order repaired and checked by reading, not compiled, not run on hardware
//
#include <hip/hip_runtime.h>

typedef __attribute__((ext_vector_type(16))) _Float16 v16h;
typedef __attribute__((ext_vector_type(8)))  float    v8f;

#define BETA 0.9f
#define B_SZ 512
#define T_SZ 200
#define I_SZ 784
#define H_SZ 128
#define O_SZ 10

union Frag16 { v16h v; unsigned u[8]; };
union Pack4  { _Float16 h[4]; uint2 u2; };

// CDNA5 async global->LDS copy (raw bytes, tracked by ASYNCcnt).
// vdst = LDS byte offset (wave-relative), addr = 64-bit global address (GV mode).
__device__ __forceinline__ void async_load_b128(unsigned lds_off, const void* gptr) {
    asm volatile("global_load_async_to_lds_b128 %0, %1, off"
                 :: "v"(lds_off), "v"(gptr) : "memory");
}
__device__ __forceinline__ void wait_async_le1() {
    asm volatile("s_wait_asynccnt 0x1" ::: "memory");
}

// ---------------------------------------------------------------------------
// Kernel 1: CUR1[n, h] = f16( X[n, :] @ W1[h, :] + b1[h] )
//   X viewed as [B*T, 784] (row n = b*T + t), W1: [128, 784]
//   CUR1 (f16) lives in d_ws: [102400, 128]  (~26 MB)
//   Software-pipelined K loop: double-buffered LDS, one barrier/iteration,
//   global loads for chunk kc+1 in flight while WMMAs run on chunk kc.
// ---------------------------------------------------------------------------
#define K_CHUNKS 25   // ceil(784 / 32)
#define LDX 36        // LDS stride (halfs): keeps 4-half groups 8B-aligned

__global__ __launch_bounds__(256) void snn_gemm1(const float* __restrict__ X,
                                                 const float* __restrict__ W1,
                                                 const float* __restrict__ b1,
                                                 _Float16* __restrict__ cur1)
{
    __shared__ _Float16 lx[2][128 * LDX];
    __shared__ _Float16 lw[2][128 * LDX];
    __shared__ float    lb1[128];

    const int tid  = threadIdx.x;
    const int lane = tid & 31;
    const int wvid = tid >> 5;     // wave 0..7 -> M tile
    const int half = lane >> 4;    // lane half
    const int l16  = lane & 15;

    const int rowbase = blockIdx.x * 128;

    // Per-thread staging coordinates: 4 quads of 4 floats per array
    const int sr[4] = { (tid + 0)   >> 3, (tid + 256) >> 3,
                        (tid + 512) >> 3, (tid + 768) >> 3 };
    const int sk = (tid & 7) << 2;  // k offset 0,4,...,28 (same for all 4 quads)

    if (tid < 128) lb1[tid] = b1[tid];

    float4 xs[4], ws4[4];

    // ---- prologue: load + stage chunk 0 ----
#pragma unroll
    for (int i = 0; i < 4; ++i) {
        xs[i]  = *(const float4*)&X[(rowbase + sr[i]) * I_SZ + sk];
        ws4[i] = *(const float4*)&W1[sr[i] * I_SZ + sk];
    }
#pragma unroll
    for (int i = 0; i < 4; ++i) {
        Pack4 px, pw;
        px.h[0] = (_Float16)xs[i].x;  px.h[1] = (_Float16)xs[i].y;
        px.h[2] = (_Float16)xs[i].z;  px.h[3] = (_Float16)xs[i].w;
        pw.h[0] = (_Float16)ws4[i].x; pw.h[1] = (_Float16)ws4[i].y;
        pw.h[2] = (_Float16)ws4[i].z; pw.h[3] = (_Float16)ws4[i].w;
        *(uint2*)&lx[0][sr[i] * LDX + sk] = px.u2;
        *(uint2*)&lw[0][sr[i] * LDX + sk] = pw.u2;
    }

    v8f zero = {};
    v8f acc[8];
#pragma unroll
    for (int nt = 0; nt < 8; ++nt) acc[nt] = zero;

    for (int kc = 0; kc < K_CHUNKS; ++kc) {
        const int cur = kc & 1;
        const int nxt = cur ^ 1;
        const bool have_next = (kc + 1 < K_CHUNKS);

        // Issue global loads for the next chunk before the barrier so they
        // are in flight during this chunk's WMMAs (zero-pad the K tail).
        if (have_next) {
            const int kb = (kc + 1) * 32;
            const int gk = kb + sk;          // 784 % 4 == 0: quads don't straddle
            if (gk < I_SZ) {
#pragma unroll
                for (int i = 0; i < 4; ++i) {
                    xs[i]  = *(const float4*)&X[(rowbase + sr[i]) * I_SZ + gk];
                    ws4[i] = *(const float4*)&W1[sr[i] * I_SZ + gk];
                }
            } else {
                float4 z4 = make_float4(0.f, 0.f, 0.f, 0.f);
#pragma unroll
                for (int i = 0; i < 4; ++i) { xs[i] = z4; ws4[i] = z4; }
            }
        }

        __syncthreads();   // lds[cur] stores from previous iteration visible

        // A fragment: 16x32 f16, row = wvid*16 + (lane&15), layout per ISA 7.12.2
        Frag16 a;
#pragma unroll
        for (int v = 0; v < 8; ++v) {
            int k0 = ((v >> 2) << 4) + (half << 3) + ((v & 3) << 1);
            a.u[v] = *(const unsigned*)&lx[cur][(wvid * 16 + l16) * LDX + k0];
        }
#pragma unroll
        for (int nt = 0; nt < 8; ++nt) {
            Frag16 bf;
#pragma unroll
            for (int v = 0; v < 8; ++v) {
                int kk = ((v >> 2) << 4) + (half << 3) + ((v & 3) << 1);
                bf.u[v] = *(const unsigned*)&lw[cur][(nt * 16 + l16) * LDX + kk];
            }
            acc[nt] = __builtin_amdgcn_wmma_f32_16x16x32_f16(
                false, a.v, false, bf.v, (short)0, acc[nt], false, false);
        }

        // Stage next chunk into the other buffer (read last iteration, so the
        // barrier above already ordered all its readers before these writes).
        if (have_next) {
#pragma unroll
            for (int i = 0; i < 4; ++i) {
                Pack4 px, pw;
                px.h[0] = (_Float16)xs[i].x;  px.h[1] = (_Float16)xs[i].y;
                px.h[2] = (_Float16)xs[i].z;  px.h[3] = (_Float16)xs[i].w;
                pw.h[0] = (_Float16)ws4[i].x; pw.h[1] = (_Float16)ws4[i].y;
                pw.h[2] = (_Float16)ws4[i].z; pw.h[3] = (_Float16)ws4[i].w;
                *(uint2*)&lx[nxt][sr[i] * LDX + sk] = px.u2;
                *(uint2*)&lw[nxt][sr[i] * LDX + sk] = pw.u2;
            }
        }
    }

    // Epilogue: D element (m,n): lane -> n = lane&15, m = v + 8*half
#pragma unroll
    for (int nt = 0; nt < 8; ++nt) {
        int col  = nt * 16 + l16;
        float bb = lb1[col];
#pragma unroll
        for (int v = 0; v < 8; ++v) {
            int row = rowbase + wvid * 16 + v + 8 * half;
            cur1[row * H_SZ + col] = (_Float16)(acc[nt][v] + bb);
        }
    }
}

// ---------------------------------------------------------------------------
// Kernel 2: sequential LIF scan. Each block owns 16 batch rows, loops t=0..199.
//   CUR1 slices streamed via double-buffered async global->LDS (B128) loads;
//   layer-2/3 GEMMs via WMMA; membranes/spikes in VGPRs, spikes via LDS f16.
// ---------------------------------------------------------------------------
#define LDS_W 132
#define LDS_S 132

__global__ __launch_bounds__(256) void snn_recur(const _Float16* __restrict__ cur1,
                                                 const float* __restrict__ W2,
                                                 const float* __restrict__ b2,
                                                 const float* __restrict__ W3,
                                                 const float* __restrict__ b3,
                                                 float* __restrict__ out)
{
    __shared__ _Float16 lw2[128 * LDS_W];
    __shared__ _Float16 lw3[16 * LDS_W];
    __shared__ _Float16 ls1[16 * LDS_S];
    __shared__ _Float16 ls2[16 * LDS_S];
    __shared__ _Float16 lcur[2][256 * 8];   // double-buffered CUR1 slices
    __shared__ float    lb2[128];
    __shared__ float    lb3[16];

    const int tid   = threadIdx.x;
    const int lane  = tid & 31;
    const int wvid  = tid >> 5;    // wave 0..7 -> layer-2 N tile
    const int half  = lane >> 4;
    const int l16   = lane & 15;
    const int bbase = blockIdx.x * 16;

    // Stage W2 (128x128), W3 (padded to 16x128), biases into LDS as f16/f32
    for (int i = 0; i < 64; ++i) {
        int e = tid + i * 256;          // 0..16383
        int r = e >> 7, k = e & 127;
        lw2[r * LDS_W + k] = (_Float16)W2[r * H_SZ + k];
    }
#pragma unroll
    for (int i = 0; i < 8; ++i) {
        int e = tid + i * 256;          // 0..2047
        int r = e >> 7, k = e & 127;
        lw3[r * LDS_W + k] = (r < O_SZ) ? (_Float16)W3[r * H_SZ + k] : (_Float16)0.f;
    }
    if (tid < 128) lb2[tid] = b2[tid];
    if (tid < 16)  lb3[tid] = (tid < O_SZ) ? b3[tid] : 0.f;

    // Per-thread persistent LIF state (THR == 1.0)
    float m1r[8], s1r[8], m2r[8], s2r[8], m3r[8], s3r[8];
#pragma unroll
    for (int j = 0; j < 8; ++j) {
        m1r[j] = 0.f; s1r[j] = 0.f; m2r[j] = 0.f;
        s2r[j] = 0.f; m3r[j] = 0.f; s3r[j] = 0.f;
    }

    const int r1 = tid >> 4;            // batch row within tile (0..15)
    const int h1 = (tid & 15) * 8;      // 8 hidden units per thread
    const _Float16* gp = cur1 + ((bbase + r1) * T_SZ) * H_SZ + h1;

    // LDS byte offsets of this thread's 16-byte double-buffer slots
    const unsigned lds0 = (unsigned)(uintptr_t)&lcur[0][tid * 8];
    const unsigned lds1 = (unsigned)(uintptr_t)&lcur[1][tid * 8];

    // Pre-issue async copy of step 0 slice (ASYNCcnt -> 1)
    async_load_b128(lds0, gp);

    __syncthreads();

    for (int t = 0; t < T_SZ; ++t) {
        // Issue async copy for the next step into the other buffer, then wait
        // for the older one (ASYNCcnt <= 1). Each lane reads back only the
        // bytes it loaded itself, so per-wave ASYNCcnt ordering suffices.
        int tn = (t + 1 < T_SZ) ? (t + 1) : t;
        async_load_b128((t & 1) ? lds0 : lds1, gp + tn * H_SZ);
        wait_async_le1();

        // ---- Layer 1 LIF ----
        const _Float16* myc = &lcur[t & 1][tid * 8];
#pragma unroll
        for (int j = 0; j < 8; ++j) {
            float m = BETA * m1r[j] + (float)myc[j] - s1r[j];
            float s = (m > 1.0f) ? 1.0f : 0.0f;
            m1r[j] = m; s1r[j] = s;
            ls1[r1 * LDS_S + h1 + j] = (_Float16)s;
        }
        __syncthreads();

        // ---- Layer 2: S1[16x128] @ W2^T, one 16x16 N-tile per wave ----
        {
            v8f acc = {};
#pragma unroll
            for (int kc = 0; kc < 4; ++kc) {
                Frag16 a, bf;
#pragma unroll
                for (int v = 0; v < 8; ++v) {
                    int k0 = kc * 32 + ((v >> 2) << 4) + (half << 3) + ((v & 3) << 1);
                    a.u[v] = *(const unsigned*)&ls1[l16 * LDS_S + k0];
                    int kk = kc * 32 + ((v >> 2) << 4) + (half << 3) + ((v & 3) << 1);
                    bf.u[v] = *(const unsigned*)&lw2[(wvid * 16 + l16) * LDS_W + kk];
                }
                acc = __builtin_amdgcn_wmma_f32_16x16x32_f16(
                    false, a.v, false, bf.v, (short)0, acc, false, false);
            }
            int col  = wvid * 16 + l16;
            float bb = lb2[col];
#pragma unroll
            for (int v = 0; v < 8; ++v) {
                float m = BETA * m2r[v] + (acc[v] + bb) - s2r[v];
                float s = (m > 1.0f) ? 1.0f : 0.0f;
                m2r[v] = m; s2r[v] = s;
                ls2[(v + 8 * half) * LDS_S + col] = (_Float16)s;
            }
        }
        __syncthreads();

        // ---- Layer 3: S2[16x128] @ W3^T (wave 0 only; wave-uniform branch) ----
        if (wvid == 0) {
            v8f acc = {};
#pragma unroll
            for (int kc = 0; kc < 4; ++kc) {
                Frag16 a, bf;
#pragma unroll
                for (int v = 0; v < 8; ++v) {
                    int k0 = kc * 32 + ((v >> 2) << 4) + (half << 3) + ((v & 3) << 1);
                    a.u[v] = *(const unsigned*)&ls2[l16 * LDS_S + k0];
                    int kk = kc * 32 + ((v >> 2) << 4) + (half << 3) + ((v & 3) << 1);
                    bf.u[v] = *(const unsigned*)&lw3[l16 * LDS_W + kk];
                }
                acc = __builtin_amdgcn_wmma_f32_16x16x32_f16(
                    false, a.v, false, bf.v, (short)0, acc, false, false);
            }
            float bb = lb3[l16];
#pragma unroll
            for (int v = 0; v < 8; ++v) {
                float m = BETA * m3r[v] + (acc[v] + bb) - s3r[v];
                float s = (m > 1.0f) ? 1.0f : 0.0f;
                m3r[v] = m; s3r[v] = s;
                if (l16 < O_SZ) {
                    int brow = bbase + v + 8 * half;
                    out[t * (B_SZ * O_SZ) + brow * O_SZ + l16] = s;
                }
            }
        }
        __syncthreads();
    }
}

// ---------------------------------------------------------------------------
extern "C" void kernel_launch(void* const* d_in, const int* in_sizes, int n_in,
                              void* d_out, int out_size, void* d_ws, size_t ws_size,
                              hipStream_t stream)
{
    const float* x  = (const float*)d_in[0];   // [512, 200, 784]
    const float* W1 = (const float*)d_in[1];   // [128, 784]
    const float* b1 = (const float*)d_in[2];   // [128]
    const float* W2 = (const float*)d_in[3];   // [128, 128]
    const float* b2 = (const float*)d_in[4];   // [128]
    const float* W3 = (const float*)d_in[5];   // [10, 128]
    const float* b3 = (const float*)d_in[6];   // [10]
    float* out = (float*)d_out;                // [200, 512, 10]
    _Float16* cur1 = (_Float16*)d_ws;          // [102400, 128] f16 (~26 MB)

    // Phase 1: time-parallel input GEMM (86% of FLOPs), 800 tiles of 128x128
    snn_gemm1<<<dim3(800), dim3(256), 0, stream>>>(x, W1, b1, cur1);
    // Phase 2: batch-parallel sequential LIF scan (32 independent batch tiles)
    snn_recur<<<dim3(32), dim3(256), 0, stream>>>(cur1, W2, b2, W3, b3, out);
}
